// InfiniAttention_7456063226351
// MI455X (gfx1250) — matrix operand order repaired and checked by reading, not compiled, not run on hardware
//
#include <hip/hip_runtime.h>

// ---------------------------------------------------------------------------
// InfiniAttention forward for MI455X (gfx1250, wave32, WMMA bf16 path).
// B=1, S=2048, HIDDEN=2048, 16 heads x 128 head_dim.
// All matmuls run through v_wmma_f32_16x16x32_bf16 (fp32 accumulate).
// GEMM core: 128x128 tile, 8 waves as 4x2 (each wave 32 rows x 64 cols ->
// 12 ds_load_b128 per 8 WMMAs), double-buffered LDS. Staging uses the CDNA5
// async global->LDS DMA when the builtin exists, else register staging.
// ---------------------------------------------------------------------------

#define SEQ 2048
#define HID 2048
#define NH  16
#define HD  128

#if __has_builtin(__builtin_amdgcn_global_load_async_to_lds_b128) && \
    __has_builtin(__builtin_amdgcn_s_wait_asynccnt)
#define HAVE_ASYNC_LDS 1
#else
#define HAVE_ASYNC_LDS 0
#endif

typedef int v4i_vec __attribute__((vector_size(16)));

#if HAVE_ASYNC_LDS
// Probe-confirmed parameter type: pointer to <4 x i32> (generic AS).
#define ASYNC_CP16(gsrc, ldst) \
  __builtin_amdgcn_global_load_async_to_lds_b128((v4i_vec*)(gsrc), (v4i_vec*)(ldst), 0, 0)
#endif

typedef __attribute__((ext_vector_type(16))) __bf16 v16bf;
typedef __attribute__((ext_vector_type(8)))  float  v8f;

// A-fragment (16x32 bf16): lane holds K = [hi*8 .. hi*8+7] and [hi*8+16 .. hi*8+23]
// B-fragment (32x16 bf16): lane holds K = [hi*8 .. hi*8+7] and [hi*8+16 .. hi*8+23], as A
__device__ __forceinline__ v16bf frag2(const __bf16* p0, const __bf16* p1) {
  union { uint4 u[2]; v16bf v; } f;
  f.u[0] = *reinterpret_cast<const uint4*>(p0);
  f.u[1] = *reinterpret_cast<const uint4*>(p1);
  return f.v;
}

__device__ __forceinline__ uint4 pack8(const v8f a) {
  union { __bf16 h[8]; uint4 u; } p;
#pragma unroll
  for (int i = 0; i < 8; ++i) p.h[i] = (__bf16)a[i];
  return p.u;
}

#define LSTR 40                 // padded LDS row stride (bf16): 80B = 5x16B
#define TILE_ELEMS (128 * LSTR) // one 128x32 staged tile

// C = A[128 x K] * B[128 x K]^T, both row-major bf16 (NT GEMM).
// 256 threads / 8 waves arranged 4x2: wave (rw,cw) owns rows [rw*32,+32),
// cols [cw*64,+64). acc[mt][t]: rows rw*32+mt*16+hi*8+r, col cw*64+t*16+(lane&15).
// As/Bs must each hold 2*TILE_ELEMS (double buffered).
__device__ __forceinline__ void gemm_nt_128x128(
    const __bf16* __restrict__ A, int lda,
    const __bf16* __restrict__ B, int ldb,
    int K, __bf16* As, __bf16* Bs, v8f acc[2][4]) {
  const int tid  = threadIdx.x;
  const int wave = tid >> 5;
  const int lane = tid & 31;
  const int lr   = lane & 15;
  const int hi   = lane >> 4;
  const int rw   = wave >> 1;  // 0..3
  const int cw   = wave & 1;   // 0..1
  // Each thread stages 2 of the 512 16B-chunks per tile (chunk i: row i>>2, col (i&3)*8).
  const int r0 = tid >> 2;
  const int r1 = (tid + 256) >> 2;
  const int c0 = (tid & 3) * 8;
  const int KT = K >> 5;

#if HAVE_ASYNC_LDS
  // Prologue: tile 0 -> buffer 0 via async DMA.
  ASYNC_CP16(A + (size_t)r0 * lda + c0, As + r0 * LSTR + c0);
  ASYNC_CP16(A + (size_t)r1 * lda + c0, As + r1 * LSTR + c0);
  ASYNC_CP16(B + (size_t)r0 * ldb + c0, Bs + r0 * LSTR + c0);
  ASYNC_CP16(B + (size_t)r1 * ldb + c0, Bs + r1 * LSTR + c0);
  __builtin_amdgcn_s_wait_asynccnt(0);
  for (int kt = 0; kt < KT; ++kt) {
    const int cur = (kt & 1) * TILE_ELEMS;
    const int nxt = ((kt + 1) & 1) * TILE_ELEMS;
    const int kn  = (kt + 1) << 5;
    __syncthreads();  // buffer `cur` published to all waves
    if (kt + 1 < KT) {  // DMA next tile into the other buffer while computing
      ASYNC_CP16(A + (size_t)r0 * lda + kn + c0, As + nxt + r0 * LSTR + c0);
      ASYNC_CP16(A + (size_t)r1 * lda + kn + c0, As + nxt + r1 * LSTR + c0);
      ASYNC_CP16(B + (size_t)r0 * ldb + kn + c0, Bs + nxt + r0 * LSTR + c0);
      ASYNC_CP16(B + (size_t)r1 * ldb + kn + c0, Bs + nxt + r1 * LSTR + c0);
    }
    const __bf16* ab = As + cur + (rw * 32 + lr) * LSTR + hi * 8;
    const v16bf a0 = frag2(ab, ab + 16);
    const v16bf a1 = frag2(ab + 16 * LSTR, ab + 16 * LSTR + 16);
    v16bf b[4];
#pragma unroll
    for (int t = 0; t < 4; ++t) {
      const __bf16* brow = Bs + cur + (cw * 64 + t * 16 + lr) * LSTR + hi * 8;
      b[t] = frag2(brow, brow + 16);
    }
#pragma unroll
    for (int t = 0; t < 4; ++t) {
      acc[0][t] = __builtin_amdgcn_wmma_f32_16x16x32_bf16(
          false, a0, false, b[t], (short)0, acc[0][t], false, false);
      acc[1][t] = __builtin_amdgcn_wmma_f32_16x16x32_bf16(
          false, a1, false, b[t], (short)0, acc[1][t], false, false);
    }
    __builtin_amdgcn_s_wait_asynccnt(0);  // next buffer resident before top barrier
  }
#else
  uint4 ra0, ra1, rb0, rb1;
  ra0 = *reinterpret_cast<const uint4*>(A + (size_t)r0 * lda + c0);
  ra1 = *reinterpret_cast<const uint4*>(A + (size_t)r1 * lda + c0);
  rb0 = *reinterpret_cast<const uint4*>(B + (size_t)r0 * ldb + c0);
  rb1 = *reinterpret_cast<const uint4*>(B + (size_t)r1 * ldb + c0);
  *reinterpret_cast<uint4*>(As + r0 * LSTR + c0) = ra0;
  *reinterpret_cast<uint4*>(As + r1 * LSTR + c0) = ra1;
  *reinterpret_cast<uint4*>(Bs + r0 * LSTR + c0) = rb0;
  *reinterpret_cast<uint4*>(Bs + r1 * LSTR + c0) = rb1;
  for (int kt = 0; kt < KT; ++kt) {
    const int cur = (kt & 1) * TILE_ELEMS;
    const int nxt = ((kt + 1) & 1) * TILE_ELEMS;
    const int kn  = (kt + 1) << 5;
    if (kt + 1 < KT) {
      ra0 = *reinterpret_cast<const uint4*>(A + (size_t)r0 * lda + kn + c0);
      ra1 = *reinterpret_cast<const uint4*>(A + (size_t)r1 * lda + kn + c0);
      rb0 = *reinterpret_cast<const uint4*>(B + (size_t)r0 * ldb + kn + c0);
      rb1 = *reinterpret_cast<const uint4*>(B + (size_t)r1 * ldb + kn + c0);
      if (kt + 2 < KT) {
        __builtin_prefetch(A + (size_t)r0 * lda + kn + 32 + c0, 0, 3);
        __builtin_prefetch(B + (size_t)r0 * ldb + kn + 32 + c0, 0, 3);
      }
    }
    __syncthreads();
    const __bf16* ab = As + cur + (rw * 32 + lr) * LSTR + hi * 8;
    const v16bf a0 = frag2(ab, ab + 16);
    const v16bf a1 = frag2(ab + 16 * LSTR, ab + 16 * LSTR + 16);
    v16bf b[4];
#pragma unroll
    for (int t = 0; t < 4; ++t) {
      const __bf16* brow = Bs + cur + (cw * 64 + t * 16 + lr) * LSTR + hi * 8;
      b[t] = frag2(brow, brow + 16);
    }
#pragma unroll
    for (int t = 0; t < 4; ++t) {
      acc[0][t] = __builtin_amdgcn_wmma_f32_16x16x32_bf16(
          false, a0, false, b[t], (short)0, acc[0][t], false, false);
      acc[1][t] = __builtin_amdgcn_wmma_f32_16x16x32_bf16(
          false, a1, false, b[t], (short)0, acc[1][t], false, false);
    }
    if (kt + 1 < KT) {
      *reinterpret_cast<uint4*>(As + nxt + r0 * LSTR + c0) = ra0;
      *reinterpret_cast<uint4*>(As + nxt + r1 * LSTR + c0) = ra1;
      *reinterpret_cast<uint4*>(Bs + nxt + r0 * LSTR + c0) = rb0;
      *reinterpret_cast<uint4*>(Bs + nxt + r1 * LSTR + c0) = rb1;
    }
  }
#endif
}

// ---------------------------------------------------------------------------
__global__ __launch_bounds__(256) void f32_to_bf16_k(const float* __restrict__ s,
                                                     __bf16* __restrict__ d, int n) {
  for (int i = blockIdx.x * 256 + threadIdx.x; i < n; i += gridDim.x * 256)
    d[i] = (__bf16)s[i];
}

// QKV: z=0 -> Qraw f32, z=1 -> Kraw f32, z=2 -> V stored transposed bf16 VT[col][s]
__global__ __launch_bounds__(256) void gemm_qkv_k(
    const __bf16* __restrict__ X, const __bf16* __restrict__ Wq,
    const __bf16* __restrict__ Wk, const __bf16* __restrict__ Wv,
    float* __restrict__ Qraw, float* __restrict__ Kraw, __bf16* __restrict__ VT) {
  const int ib = blockIdx.x, nb = blockIdx.y, z = blockIdx.z;
  const __bf16* W = (z == 0) ? Wq : (z == 1) ? Wk : Wv;
  __shared__ __bf16 As[2 * TILE_ELEMS];
  __shared__ __bf16 Bs[2 * TILE_ELEMS];
  v8f acc[2][4] = {};
  gemm_nt_128x128(X + (size_t)ib * 128 * HID, HID,
                  W + (size_t)nb * 128 * HID, HID, HID, As, Bs, acc);
  const int tid = threadIdx.x, wave = tid >> 5, lane = tid & 31;
  const int lr = lane & 15, hi = lane >> 4, rw = wave >> 1, cw = wave & 1;
  if (z < 2) {
    float* R = z ? Kraw : Qraw;
#pragma unroll
    for (int mt = 0; mt < 2; ++mt) {
      const int row0 = ib * 128 + rw * 32 + mt * 16 + hi * 8;
#pragma unroll
      for (int t = 0; t < 4; ++t) {
        const int col = nb * 128 + cw * 64 + t * 16 + lr;
#pragma unroll
        for (int r = 0; r < 8; ++r)
          R[(size_t)(row0 + r) * HID + col] = acc[mt][t][r];
      }
    }
  } else {
#pragma unroll
    for (int mt = 0; mt < 2; ++mt) {
      const int row0 = ib * 128 + rw * 32 + mt * 16 + hi * 8;
#pragma unroll
      for (int t = 0; t < 4; ++t) {
        const int col = nb * 128 + cw * 64 + t * 16 + lr;  // col == h*128 + e
        *reinterpret_cast<uint4*>(VT + (size_t)col * SEQ + row0) = pack8(acc[mt][t]);
      }
    }
  }
}

// RoPE over full hidden dim + ELU feature. isK: write k_feat transposed [col][s].
__global__ __launch_bounds__(256) void rope_feat_k(
    const float* __restrict__ Raw, __bf16* __restrict__ Rot,
    __bf16* __restrict__ featRM, __bf16* __restrict__ featT, int isK) {
  const int id = blockIdx.x * 256 + threadIdx.x;
  if (id >= SEQ * (HID / 2)) return;
  const int s = id >> 10, j = id & 1023;
  const float inv = __expf(-(2.f * (float)j / (float)HID) * 9.210340371976184f); // ln(1e4)
  float sn, cs;
  __sincosf((float)s * inv, &sn, &cs);
  const float q1 = Raw[(size_t)s * HID + j];
  const float q2 = Raw[(size_t)s * HID + j + 1024];
  const float r1 = q1 * cs - q2 * sn;
  const float r2 = q2 * cs + q1 * sn;
  Rot[(size_t)s * HID + j]        = (__bf16)r1;
  Rot[(size_t)s * HID + j + 1024] = (__bf16)r2;
  const float f1 = (r1 > 0.f) ? (r1 + 1.f) : __expf(r1);  // elu(x)+1
  const float f2 = (r2 > 0.f) ? (r2 + 1.f) : __expf(r2);
  if (isK) {
    featT[(size_t)j * SEQ + s]          = (__bf16)f1;
    featT[(size_t)(j + 1024) * SEQ + s] = (__bf16)f2;
  } else {
    featRM[(size_t)s * HID + j]        = (__bf16)f1;
    featRM[(size_t)s * HID + j + 1024] = (__bf16)f2;
  }
}

__global__ __launch_bounds__(256) void norm_k(const __bf16* __restrict__ kfT,
                                              float* __restrict__ nrm) {
  const int row = blockIdx.x * 256 + threadIdx.x;  // 0..2047 == h*128 + d
  if (row >= HID) return;
  float acc = 0.f;
  for (int s = 0; s < SEQ; ++s) acc += (float)kfT[(size_t)row * SEQ + s];
  nrm[row] = acc;
}

__global__ __launch_bounds__(256) void denom_k(const __bf16* __restrict__ qf,
                                               const float* __restrict__ nrm,
                                               float* __restrict__ den) {
  const int id = blockIdx.x * 256 + threadIdx.x;  // 0..NH*SEQ
  if (id >= NH * SEQ) return;
  const int h = id >> 11, s = id & (SEQ - 1);
  float acc = 0.f;
  for (int d = 0; d < HD; ++d)
    acc += (float)qf[(size_t)s * HID + h * HD + d] * nrm[h * HD + d];
  den[(size_t)h * SEQ + s] = acc;
}

// memT[h][e][d] = sum_s v[s,e] * k_feat[s,d]  (== memory[d][e] transposed)
__global__ __launch_bounds__(256) void memT_k(const __bf16* __restrict__ VT,
                                              const __bf16* __restrict__ kfT,
                                              __bf16* __restrict__ memT) {
  const int h = blockIdx.x;
  __shared__ __bf16 As[2 * TILE_ELEMS];
  __shared__ __bf16 Bs[2 * TILE_ELEMS];
  v8f acc[2][4] = {};
  gemm_nt_128x128(VT + (size_t)h * HD * SEQ, SEQ,
                  kfT + (size_t)h * HD * SEQ, SEQ, SEQ, As, Bs, acc);
  const int tid = threadIdx.x, wave = tid >> 5, lane = tid & 31;
  const int lr = lane & 15, hi = lane >> 4, rw = wave >> 1, cw = wave & 1;
#pragma unroll
  for (int mt = 0; mt < 2; ++mt)
#pragma unroll
    for (int t = 0; t < 4; ++t) {
      const int d = cw * 64 + t * 16 + lr;
#pragma unroll
      for (int r = 0; r < 8; ++r) {
        const int e = rw * 32 + mt * 16 + hi * 8 + r;
        memT[((size_t)h * HD + e) * HD + d] = (__bf16)acc[mt][t][r];
      }
    }
}

// mem_scaled[s][h*128+e] = sigmoid(gate_h) * (q_feat_s . memory[:,e]) / denom[h][s]
__global__ __launch_bounds__(256) void memout_k(const __bf16* __restrict__ qf,
                                                const __bf16* __restrict__ memT,
                                                const float* __restrict__ den,
                                                const float* __restrict__ gate,
                                                float* __restrict__ msc) {
  const int sb = blockIdx.x, h = blockIdx.y;
  __shared__ __bf16 As[2 * TILE_ELEMS];
  __shared__ __bf16 Bs[2 * TILE_ELEMS];
  v8f acc[2][4] = {};
  gemm_nt_128x128(qf + (size_t)sb * 128 * HID + h * HD, HID,
                  memT + (size_t)h * HD * HD, HD, HD, As, Bs, acc);
  const int tid = threadIdx.x, wave = tid >> 5, lane = tid & 31;
  const int lr = lane & 15, hi = lane >> 4, rw = wave >> 1, cw = wave & 1;
  const float g = 1.f / (1.f + __expf(-gate[h]));
#pragma unroll
  for (int mt = 0; mt < 2; ++mt)
#pragma unroll
    for (int t = 0; t < 4; ++t) {
      const int e = cw * 64 + t * 16 + lr;
#pragma unroll
      for (int r = 0; r < 8; ++r) {
        const int s = sb * 128 + rw * 32 + mt * 16 + hi * 8 + r;
        msc[(size_t)s * HID + h * HD + e] = g * acc[mt][t][r] / den[(size_t)h * SEQ + s];
      }
    }
}

// Flash attention per (q-block, head) + gated combine with mem path.
__global__ __launch_bounds__(256) void flash_k(
    const __bf16* __restrict__ Qr, const __bf16* __restrict__ Kr,
    const __bf16* __restrict__ VT, const float* __restrict__ msc,
    const float* __restrict__ gate, __bf16* __restrict__ comb) {
  const int sb = blockIdx.x, h = blockIdx.y;
  __shared__ __bf16 Ks[128 * 136];       // K block  [kv][d]   (+pad)
  __shared__ __bf16 VTs[128 * 136];      // VT block [e][kv]   (+pad)
  __shared__ __bf16 Ps[8 * 16 * 136];    // per-wave P [q][kv] (+pad)
  const int tid = threadIdx.x, wave = tid >> 5, lane = tid & 31;
  const int lr = lane & 15, hi = lane >> 4;
  const float scale = 0.08838834764831845f;  // 1/sqrt(128)

  // Q fragments persist in registers across all kv blocks.
  v16bf aQ[4];
  {
    const __bf16* qb = Qr + (size_t)(sb * 128 + wave * 16 + lr) * HID + h * HD + hi * 8;
#pragma unroll
    for (int kd = 0; kd < 4; ++kd) aQ[kd] = frag2(qb + kd * 32, qb + kd * 32 + 16);
  }
  float mi[8], li[8];
  v8f O[8] = {};
#pragma unroll
  for (int r = 0; r < 8; ++r) { mi[r] = -1e30f; li[r] = 0.f; }
  __bf16* psb = Ps + wave * 16 * 136;

  for (int j = 0; j < 16; ++j) {
    __syncthreads();
#if HAVE_ASYNC_LDS
#pragma unroll 4
    for (int i = tid; i < 128 * 16; i += 256) {
      const int r = i >> 4, c = (i & 15) * 8;
      ASYNC_CP16(Kr + (size_t)(j * 128 + r) * HID + h * HD + c, Ks + r * 136 + c);
      ASYNC_CP16(VT + (size_t)(h * HD + r) * SEQ + j * 128 + c, VTs + r * 136 + c);
    }
    __builtin_amdgcn_s_wait_asynccnt(0);
#else
#pragma unroll 4
    for (int i = tid; i < 128 * 16; i += 256) {
      const int r = i >> 4, c = (i & 15) * 8;
      *reinterpret_cast<uint4*>(Ks + r * 136 + c) =
          *reinterpret_cast<const uint4*>(Kr + (size_t)(j * 128 + r) * HID + h * HD + c);
      *reinterpret_cast<uint4*>(VTs + r * 136 + c) =
          *reinterpret_cast<const uint4*>(VT + (size_t)(h * HD + r) * SEQ + j * 128 + c);
    }
#endif
    if (j + 1 < 16) {  // near prefetch of next kv block (two 128B halves per row)
      const int pr = tid >> 1, pc = (tid & 1) * 64;
      __builtin_prefetch(Kr + (size_t)((j + 1) * 128 + pr) * HID + h * HD + pc, 0, 3);
      __builtin_prefetch(VT + (size_t)(h * HD + pr) * SEQ + (j + 1) * 128 + pc, 0, 3);
    }
    __syncthreads();

    // S = Q * K^T (128-deep reduction over head_dim); batch B frags per k-step.
    v8f S[8] = {};
#pragma unroll
    for (int kd = 0; kd < 4; ++kd) {
      v16bf b[8];
#pragma unroll
      for (int t = 0; t < 8; ++t) {
        const __bf16* krow = Ks + (t * 16 + lr) * 136 + kd * 32 + hi * 8;
        b[t] = frag2(krow, krow + 16);
      }
#pragma unroll
      for (int t = 0; t < 8; ++t)
        S[t] = __builtin_amdgcn_wmma_f32_16x16x32_bf16(
            false, aQ[kd], false, b[t], (short)0, S[t], false, false);
    }

    // Online softmax. Lane owns rows hi*8 + r; reduce across its 16-lane half.
#pragma unroll
    for (int r = 0; r < 8; ++r) {
      float m = -1e30f;
#pragma unroll
      for (int t = 0; t < 8; ++t) m = fmaxf(m, S[t][r]);
      m *= scale;
      for (int msk = 1; msk < 16; msk <<= 1) m = fmaxf(m, __shfl_xor(m, msk, 32));
      const float mn = fmaxf(mi[r], m);
      float sum = 0.f;
#pragma unroll
      for (int t = 0; t < 8; ++t) {
        const float p = __expf(S[t][r] * scale - mn);
        S[t][r] = p;
        sum += p;
      }
      for (int msk = 1; msk < 16; msk <<= 1) sum += __shfl_xor(sum, msk, 32);
      const float alpha = __expf(mi[r] - mn);
      li[r] = li[r] * alpha + sum;
      mi[r] = mn;
#pragma unroll
      for (int t = 0; t < 8; ++t) O[t][r] *= alpha;
    }

    // Re-shape P (C-layout) -> A fragments through this wave's LDS slice.
#pragma unroll
    for (int t = 0; t < 8; ++t)
#pragma unroll
      for (int r = 0; r < 8; ++r)
        psb[(hi * 8 + r) * 136 + t * 16 + lr] = (__bf16)S[t][r];

    // O += P * V  (B operand from transposed V: lane e holds contiguous kv)
#pragma unroll
    for (int kd = 0; kd < 4; ++kd) {
      const __bf16* prow = psb + lr * 136 + kd * 32 + hi * 8;
      const v16bf a = frag2(prow, prow + 16);
      v16bf b[8];
#pragma unroll
      for (int t = 0; t < 8; ++t) {
        const __bf16* vrow = VTs + (t * 16 + lr) * 136 + kd * 32 + hi * 8;
        b[t] = frag2(vrow, vrow + 16);
      }
#pragma unroll
      for (int t = 0; t < 8; ++t)
        O[t] = __builtin_amdgcn_wmma_f32_16x16x32_bf16(
            false, a, false, b[t], (short)0, O[t], false, false);
    }
  }

  // combined = g*mem_out/denom + (1-g)*attn  (mem term pre-scaled in msc)
  const float og = 1.f - 1.f / (1.f + __expf(-gate[h]));
#pragma unroll
  for (int t = 0; t < 8; ++t) {
    const int col = h * HD + t * 16 + lr;
#pragma unroll
    for (int r = 0; r < 8; ++r) {
      const int s = sb * 128 + wave * 16 + hi * 8 + r;
      const float attn = O[t][r] / li[r];
      comb[(size_t)s * HID + col] = (__bf16)(msc[(size_t)s * HID + col] + og * attn);
    }
  }
}

__global__ __launch_bounds__(256) void gemm_out_k(const __bf16* __restrict__ comb,
                                                  const __bf16* __restrict__ Wo,
                                                  float* __restrict__ out) {
  const int ib = blockIdx.x, nb = blockIdx.y;
  __shared__ __bf16 As[2 * TILE_ELEMS];
  __shared__ __bf16 Bs[2 * TILE_ELEMS];
  v8f acc[2][4] = {};
  gemm_nt_128x128(comb + (size_t)ib * 128 * HID, HID,
                  Wo + (size_t)nb * 128 * HID, HID, HID, As, Bs, acc);
  const int tid = threadIdx.x, wave = tid >> 5, lane = tid & 31;
  const int lr = lane & 15, hi = lane >> 4, rw = wave >> 1, cw = wave & 1;
#pragma unroll
  for (int mt = 0; mt < 2; ++mt) {
    const int row0 = ib * 128 + rw * 32 + mt * 16 + hi * 8;
#pragma unroll
    for (int t = 0; t < 4; ++t) {
      const int col = nb * 128 + cw * 64 + t * 16 + lr;
#pragma unroll
      for (int r = 0; r < 8; ++r)
        out[(size_t)(row0 + r) * HID + col] = acc[mt][t][r];
    }
  }
}

// ---------------------------------------------------------------------------
extern "C" void kernel_launch(void* const* d_in, const int* in_sizes, int n_in,
                              void* d_out, int out_size, void* d_ws, size_t ws_size,
                              hipStream_t stream) {
  (void)in_sizes; (void)n_in; (void)out_size; (void)ws_size;
  const float* hs   = (const float*)d_in[0];
  const float* Wq   = (const float*)d_in[1];
  const float* Wk   = (const float*)d_in[2];
  const float* Wv   = (const float*)d_in[3];
  const float* Wo   = (const float*)d_in[4];
  const float* gate = (const float*)d_in[5];
  float* out = (float*)d_out;

  constexpr size_t NELT = (size_t)SEQ * HID;  // 4Mi elements
  char* ws = (char*)d_ws;
  size_t off = 0;
  auto alloc = [&](size_t bytes) -> char* {
    char* p = ws + off;
    off += (bytes + 255) & ~(size_t)255;
    return p;
  };
  __bf16* Xb   = (__bf16*)alloc(NELT * 2);
  __bf16* Wqb  = (__bf16*)alloc(NELT * 2);
  __bf16* Wkb  = (__bf16*)alloc(NELT * 2);
  __bf16* Wvb  = (__bf16*)alloc(NELT * 2);
  __bf16* Wob  = (__bf16*)alloc(NELT * 2);
  float*  Qraw = (float*)alloc(NELT * 4);
  float*  Kraw = (float*)alloc(NELT * 4);
  __bf16* Qr   = (__bf16*)alloc(NELT * 2);
  __bf16* Kr   = (__bf16*)alloc(NELT * 2);
  __bf16* qfe  = (__bf16*)alloc(NELT * 2);   // q_feat row-major
  __bf16* kfT  = (__bf16*)alloc(NELT * 2);   // k_feat transposed [col][s]
  __bf16* VT   = (__bf16*)alloc(NELT * 2);   // V transposed [col][s]
  float*  nrm  = (float*)alloc((size_t)HID * 4);
  __bf16* memT = (__bf16*)alloc((size_t)NH * HD * HD * 2);
  float*  den  = (float*)alloc((size_t)NH * SEQ * 4);
  float*  msc  = (float*)alloc(NELT * 4);    // g * mem_out / denom, row-major f32
  __bf16* comb = (__bf16*)alloc(NELT * 2);

  // 1) convert to bf16
  f32_to_bf16_k<<<2048, 256, 0, stream>>>(hs, Xb, (int)NELT);
  f32_to_bf16_k<<<2048, 256, 0, stream>>>(Wq, Wqb, (int)NELT);
  f32_to_bf16_k<<<2048, 256, 0, stream>>>(Wk, Wkb, (int)NELT);
  f32_to_bf16_k<<<2048, 256, 0, stream>>>(Wv, Wvb, (int)NELT);
  f32_to_bf16_k<<<2048, 256, 0, stream>>>(Wo, Wob, (int)NELT);
  // 2) QKV projections (WMMA)
  gemm_qkv_k<<<dim3(16, 16, 3), 256, 0, stream>>>(Xb, Wqb, Wkb, Wvb, Qraw, Kraw, VT);
  // 3) RoPE + ELU features
  rope_feat_k<<<(SEQ * (HID / 2)) / 256, 256, 0, stream>>>(Qraw, Qr, qfe, kfT, 0);
  rope_feat_k<<<(SEQ * (HID / 2)) / 256, 256, 0, stream>>>(Kraw, Kr, qfe, kfT, 1);
  // 4) compressive-memory terms
  norm_k<<<HID / 256, 256, 0, stream>>>(kfT, nrm);
  denom_k<<<(NH * SEQ) / 256, 256, 0, stream>>>(qfe, nrm, den);
  memT_k<<<NH, 256, 0, stream>>>(VT, kfT, memT);
  memout_k<<<dim3(16, NH), 256, 0, stream>>>(qfe, memT, den, gate, msc);
  // 5) flash attention + gated combine
  flash_k<<<dim3(16, NH), 256, 0, stream>>>(Qr, Kr, VT, msc, gate, comb);
  // 6) output projection
  gemm_out_k<<<dim3(16, 16), 256, 0, stream>>>(comb, Wob, out);
}
